// MXMNet_51616916964063
// MI455X (gfx1250) — matrix operand order repaired and checked by reading, not compiled, not run on hardware
//
#include <hip/hip_runtime.h>

typedef unsigned short u16;
typedef __bf16 bf16;
typedef __attribute__((ext_vector_type(8)))  bf16  bf16x8;
typedef __attribute__((ext_vector_type(16))) bf16  bf16x16;
typedef __attribute__((ext_vector_type(8)))  float f32x8;
typedef __attribute__((vector_size(16)))     int   i32x4;

#define D128 128
#define AS1 __attribute__((address_space(1)))
#define AS3 __attribute__((address_space(3)))

#if defined(__AMDGCN__) && __has_builtin(__builtin_amdgcn_global_load_async_to_lds_b128) && \
    __has_builtin(__builtin_amdgcn_s_wait_asynccnt)
#define USE_ASYNC 1
#else
#define USE_ASYNC 0
#endif

// ---------------------------------------------------------------- helpers
__device__ __forceinline__ u16 f2b(float f) {
    unsigned u = __float_as_uint(f);
    u += 0x7fffu + ((u >> 16) & 1u);          // round-to-nearest-even bf16
    return (u16)(u >> 16);
}
__device__ __forceinline__ float siluf(float x) { return x / (1.f + __expf(-x)); }

// zeros of spherical Bessel j_l, l=0..6, first 6 roots
__device__ __constant__ float SBF_Z[7][6] = {
    { 3.14159265f,  6.28318531f,  9.42477796f, 12.56637061f, 15.70796327f, 18.84955592f},
    { 4.49340946f,  7.72525184f, 10.90412166f, 14.06619391f, 17.22075527f, 20.37130296f},
    { 5.76345920f,  9.09501133f, 12.32294097f, 15.51460301f, 18.68903636f, 21.85387422f},
    { 6.98793200f, 10.41711855f, 13.69802315f, 16.92362129f, 20.12180617f, 23.30424754f},
    { 8.18256145f, 11.70490715f, 15.03966471f, 18.30125596f, 21.52541773f, 24.72755635f},
    { 9.35581211f, 12.96653017f, 16.35470964f, 19.65315287f, 22.90455064f, 26.12775227f},
    {10.51283540f, 14.20739209f, 17.64797496f, 20.98346348f, 24.26276804f, 27.50786841f}};

__device__ float jl_dev(int l, float x) {
    float j0 = __sinf(x) / x;
    if (l == 0) return j0;
    float j1 = __sinf(x) / (x * x) - __cosf(x) / x;
    float jm = j0, jc = j1;
    for (int n = 1; n < l; ++n) { float t = (2.f * n + 1.f) / x * jc - jm; jm = jc; jc = t; }
    return jc;
}
__device__ __forceinline__ float env6(float x) {  // envelope, p=6
    float x5 = x * x * x * x * x;
    return (x < 1.f) ? (1.f / x - 28.f * x5 + 48.f * x5 * x - 21.f * x5 * x * x) : 0.f;
}

// ---------------------------------------------------------------- WMMA GEMM
// Y[M,N] = act(Xbf16[M,K] @ W + bias) (+ res).  K multiple of 64, N = 128.
// W is PRE-TRANSPOSED bf16: W[n*K + k].  Dual store: outF (f32) / outB (bf16).
#define ASTR 72   // padded LDS row stride (u16) -> conflict-free fragment reads
__global__ __launch_bounds__(256) void k_gemm(
    const u16* __restrict__ X, const u16* __restrict__ W,
    const float* __restrict__ bias, const float* __restrict__ res,
    float* __restrict__ outF, u16* __restrict__ outB,
    int M, int N, int K, int act)
{
    __shared__ u16 As[2][128 * ASTR];
    __shared__ u16 Bs[2][128 * ASTR];

    const int tid    = threadIdx.x;
    const int lane   = tid & 31;
    const int wave   = tid >> 5;
    const int wm     = wave >> 2;          // 0..1  (64 rows each)
    const int wn     = wave & 3;           // 0..3  (32 cols each)
    const int laneLo = lane & 15, laneHi = lane >> 4;
    const int bm0    = blockIdx.y * 128;
    const int bn0    = blockIdx.x * 128;

    f32x8 acc[4][2];
#pragma unroll
    for (int i = 0; i < 4; ++i)
#pragma unroll
        for (int j = 0; j < 2; ++j) acc[i][j] = {};

    // A tile and (pre-transposed) B tile have identical 128x64 u16 shapes:
    // 1024 16-byte chunks each, 4 per thread.
    auto loadTiles = [&](int k0, int buf) {
#pragma unroll
        for (int i = 0; i < 4; ++i) {
            int c   = tid + i * 256;
            int row = c >> 3;
            int kc  = (c & 7) << 3;
            const u16* ga = X + (size_t)(bm0 + row) * K + k0 + kc;
            const u16* gb = W + (size_t)(bn0 + row) * K + k0 + kc;
            u16* la = &As[buf][row * ASTR + kc];
            u16* lb = &Bs[buf][row * ASTR + kc];
#if USE_ASYNC
            if (bm0 + row < M) {
                __builtin_amdgcn_global_load_async_to_lds_b128(
                    (AS1 i32x4*)(i32x4*)(void*)ga, (AS3 i32x4*)(i32x4*)(void*)la, 0, 0);
                if (k0 + 128 < K)
                    __builtin_prefetch(ga + 128, 0, 1);
            }
            __builtin_amdgcn_global_load_async_to_lds_b128(
                (AS1 i32x4*)(i32x4*)(void*)gb, (AS3 i32x4*)(i32x4*)(void*)lb, 0, 0);
#else
            uint4 v = make_uint4(0u, 0u, 0u, 0u);
            if (bm0 + row < M) {
                v = *(const uint4*)ga;
                if (k0 + 128 < K) __builtin_prefetch(ga + 128, 0, 1);
            }
            *(uint4*)la = v;
            *(uint4*)lb = *(const uint4*)gb;
#endif
        }
    };

    const int nk = K >> 6;
    int buf = 0;
    loadTiles(0, 0);
    for (int t = 0; t < nk; ++t) {
#if USE_ASYNC
        __builtin_amdgcn_s_wait_asynccnt(0);   // my async writes to LDS done
#endif
        __syncthreads();                        // publish tile to all waves
        if (t + 1 < nk) loadTiles((t + 1) << 6, buf ^ 1);
#pragma unroll
        for (int kk = 0; kk < 64; kk += 32) {
            union { bf16x16 v; bf16x8 h[2]; } a[4], b[2];
#pragma unroll
            for (int mi = 0; mi < 4; ++mi) {
                int r = wm * 64 + mi * 16 + laneLo;
                const u16* base = &As[buf][r * ASTR + kk];
                a[mi].h[0] = *(const bf16x8*)(base + laneHi * 8);
                a[mi].h[1] = *(const bf16x8*)(base + 16 + laneHi * 8);
            }
#pragma unroll
            for (int ni = 0; ni < 2; ++ni) {
                int ncol = wn * 32 + ni * 16 + laneLo;
                const u16* base = &Bs[buf][ncol * ASTR + kk + laneHi * 8];
                b[ni].h[0] = *(const bf16x8*)(base);
                b[ni].h[1] = *(const bf16x8*)(base + 16);
            }
#pragma unroll
            for (int mi = 0; mi < 4; ++mi)
#pragma unroll
                for (int ni = 0; ni < 2; ++ni)
                    acc[mi][ni] = __builtin_amdgcn_wmma_f32_16x16x32_bf16(
                        false, a[mi].v, false, b[ni].v, (short)0, acc[mi][ni], false, false);
        }
        buf ^= 1;
    }

    // C layout (ISA 7.12.2): vgpr r -> row base + r + laneHi*8, col = base + laneLo
#pragma unroll
    for (int mi = 0; mi < 4; ++mi) {
        int rowBase = bm0 + wm * 64 + mi * 16 + laneHi * 8;
#pragma unroll
        for (int ni = 0; ni < 2; ++ni) {
            int col = bn0 + wn * 32 + ni * 16 + laneLo;
            float bv = bias ? bias[col] : 0.f;
#pragma unroll
            for (int r = 0; r < 8; ++r) {
                int row = rowBase + r;
                if (row < M) {
                    float v = acc[mi][ni][r] + bv;
                    if (act) v = siluf(v);
                    if (res) v += res[(size_t)row * N + col];
                    if (outF) outF[(size_t)row * N + col] = v;
                    if (outB) outB[(size_t)row * N + col] = f2b(v);
                }
            }
        }
    }
}

// ---------------------------------------------------------------- small kernels
// pack f32 [K,N] weights -> bf16 TRANSPOSED [N,Kp], K zero-padded to Kp
__global__ void k_pack_w(const float* __restrict__ s, u16* __restrict__ d, int K, int Kp, int N) {
    int i = blockIdx.x * 256 + threadIdx.x;
    if (i >= N * Kp) return;
    int n = i / Kp, k = i - n * Kp;
    d[i] = (k < K) ? f2b(s[(size_t)k * N + n]) : (u16)0;
}

__global__ void k_embed(const float* __restrict__ emb, const int* __restrict__ z,
                        float* __restrict__ h, u16* __restrict__ hb, int N) {
    int i = blockIdx.x * 256 + threadIdx.x;
    if (i >= N * D128) return;
    int n = i >> 7, c = i & 127;
    float v = emb[z[n] * D128 + c];
    h[i] = v; hb[i] = f2b(v);
}

__global__ void k_bessel(const float* __restrict__ pos, const int* __restrict__ src,
                         const int* __restrict__ dst, const float* __restrict__ freq,
                         float* __restrict__ dOut, u16* __restrict__ bes, int E) {
    int e = blockIdx.x * 256 + threadIdx.x;
    if (e >= E) return;
    int a = src[e], b = dst[e];
    float dx = pos[b * 3 + 0] - pos[a * 3 + 0];
    float dy = pos[b * 3 + 1] - pos[a * 3 + 1];
    float dz = pos[b * 3 + 2] - pos[a * 3 + 2];
    float d = sqrtf(dx * dx + dy * dy + dz * dz);
    if (dOut) dOut[e] = d;
    float x = d * 0.2f;            // /CUTOFF
    float env = env6(x);
    u16* o = bes + (size_t)e * 64;
#pragma unroll
    for (int k = 0; k < 16; ++k) o[k] = f2b(env * __sinf(freq[k] * x));
#pragma unroll
    for (int k = 16; k < 64; ++k) o[k] = 0;
}

__global__ void k_sbf_edge(const float* __restrict__ dL, float* __restrict__ sbfe, int E) {
    int e = blockIdx.x * 256 + threadIdx.x;
    if (e >= E) return;
    float xs = dL[e] * 0.2f;
    float env = env6(xs);
    float* o = sbfe + (size_t)e * 42;
    for (int l = 0; l < 7; ++l)
        for (int r = 0; r < 6; ++r) {
            float z = SBF_Z[l][r];
            float jn = jl_dev(l + 1, z);
            float norm = rsqrtf(0.5f * jn * jn);
            o[l * 6 + r] = env * norm * jl_dev(l, z * xs);
        }
}

__global__ void k_tri(const float* __restrict__ pos, const int* __restrict__ ia,
                      const int* __restrict__ ja, const int* __restrict__ ka,
                      const int* __restrict__ eidx, const float* __restrict__ sbfe,
                      u16* __restrict__ raw, int T) {
    int t = blockIdx.x * 256 + threadIdx.x;
    if (t >= T) return;
    int i = ia[t], j = ja[t], k = ka[t];
    float v1x = pos[j * 3 + 0] - pos[i * 3 + 0], v1y = pos[j * 3 + 1] - pos[i * 3 + 1],
          v1z = pos[j * 3 + 2] - pos[i * 3 + 2];
    float v2x = pos[k * 3 + 0] - pos[j * 3 + 0], v2y = pos[k * 3 + 1] - pos[j * 3 + 1],
          v2z = pos[k * 3 + 2] - pos[j * 3 + 2];
    float dp = v1x * v2x + v1y * v2y + v1z * v2z;
    float cx = v1y * v2z - v1z * v2y, cy = v1z * v2x - v1x * v2z, cz = v1x * v2y - v1y * v2x;
    float s2 = cx * cx + cy * cy + cz * cz;
    float c = dp * rsqrtf(dp * dp + s2 + 1e-30f);   // cos(atan2(|cross|,dot))
    float P[7];
    P[0] = 1.f; P[1] = c;
    for (int l = 2; l < 7; ++l) P[l] = ((2.f * l - 1.f) * c * P[l - 1] - (l - 1.f) * P[l - 2]) / l;
    const float inv4pi = 0.07957747154594767f;
    const float* se = sbfe + (size_t)eidx[t] * 42;
    u16* o = raw + (size_t)t * 64;
    for (int l = 0; l < 7; ++l) {
        float coef = sqrtf((2.f * l + 1.f) * inv4pi) * P[l];
        for (int r = 0; r < 6; ++r) o[l * 6 + r] = f2b(se[l * 6 + r] * coef);
    }
#pragma unroll
    for (int cc = 42; cc < 64; ++cc) o[cc] = 0;
}

// X3[e] = [hm[dst[e]], hm[src[e]], rbf[e]]  (bf16, 8-elem chunks)
__global__ void k_concat3(const u16* __restrict__ hm, const int* __restrict__ dst,
                          const int* __restrict__ src, const u16* __restrict__ rbf,
                          u16* __restrict__ X3, int E) {
    unsigned i = blockIdx.x * 256 + threadIdx.x;
    if (i >= (unsigned)E * 48u) return;
    int e = i / 48, c = (i % 48) * 8;
    const u16* sp;
    if (c < 128)      sp = hm  + (size_t)dst[e] * D128 + c;
    else if (c < 256) sp = hm  + (size_t)src[e] * D128 + (c - 128);
    else              sp = rbf + (size_t)e * D128 + (c - 256);
    *(uint4*)(X3 + (size_t)e * 384 + c) = *(const uint4*)sp;
}

__global__ void k_mul(float* __restrict__ a, const float* __restrict__ b, long n) {
    long i = (long)blockIdx.x * 256 + threadIdx.x;
    if (i < n) a[i] *= b[i];
}
__global__ void k_f2b(u16* __restrict__ d, const float* __restrict__ s, long n) {
    long i = (long)blockIdx.x * 256 + threadIdx.x;
    if (i < n) d[i] = f2b(s[i]);
}
__global__ void k_scatter_add(float* __restrict__ acc, const float* __restrict__ m,
                              const int* __restrict__ idx, int E) {
    long i = (long)blockIdx.x * 256 + threadIdx.x;
    if (i >= (long)E * D128) return;
    int e = i >> 7, c = i & 127;
    atomicAdd(&acc[(size_t)idx[e] * D128 + c], m[i]);
}
// acc[ji[t]] += mkj[kj[t]] * s[t]
__global__ void k_tri_gms(float* __restrict__ acc, const float* __restrict__ mkj,
                          const float* __restrict__ s, const int* __restrict__ kj,
                          const int* __restrict__ ji, int T) {
    long i = (long)blockIdx.x * 256 + threadIdx.x;
    if (i >= (long)T * D128) return;
    int t = i >> 7, c = i & 127;
    float v = mkj[(size_t)kj[t] * D128 + c] * s[i];
    atomicAdd(&acc[(size_t)ji[t] * D128 + c], v);
}
// one wave32 per node: nodeSum[n] += ym[n,:] . Wy + by
__global__ void k_dot(const float* __restrict__ ym, const float* __restrict__ Wy,
                      const float* __restrict__ by, float* __restrict__ nodeSum, int N) {
    int n    = (blockIdx.x * 256 + threadIdx.x) >> 5;
    int lane = threadIdx.x & 31;
    if (n >= N) return;
    const float* r = ym + (size_t)n * D128;
    float s = 0.f;
#pragma unroll
    for (int c = 0; c < D128; c += 32) s += r[c + lane] * Wy[c + lane];
#pragma unroll
    for (int m = 16; m; m >>= 1) s += __shfl_xor(s, m, 32);
    if (lane == 0) nodeSum[n] += s + by[0];
}
__global__ void k_mol_sum(const float* __restrict__ nodeSum, const int* __restrict__ batch,
                          float* __restrict__ out, int N) {
    int n = blockIdx.x * 256 + threadIdx.x;
    if (n >= N) return;
    atomicAdd(&out[batch[n]], nodeSum[n]);
}

// ---------------------------------------------------------------- host
extern "C" void kernel_launch(void* const* d_in, const int* in_sizes, int n_in,
                              void* d_out, int out_size, void* d_ws, size_t ws_size,
                              hipStream_t stream) {
    if (n_in < 209) return;   // expects params pytree (191 leaves) + 18 data inputs
#define FP(i) ((const float*)d_in[(i)])
#define IP(i) ((const int*)d_in[(i)])
    auto GLOB = [](int L, int o) { return 3 + L * 19 + o; };
    auto LOC  = [](int L, int o) { return 60 + L * 41 + o; };
    const int INP = 191;
    const float* pos   = FP(INP + 0);
    const int* atom_z  = IP(INP + 1);
    const int* batch   = IP(INP + 2);
    const int* src_l   = IP(INP + 3);  const int* dst_l = IP(INP + 4);
    const int* src_g   = IP(INP + 5);  const int* dst_g = IP(INP + 6);
    const int* idx_i1  = IP(INP + 7);  const int* idx_j  = IP(INP + 8);
    const int* idx_k   = IP(INP + 9);  const int* idx_kj = IP(INP + 10);
    const int* idx_ji1 = IP(INP + 11); const int* idx_i2 = IP(INP + 12);
    const int* idx_j1  = IP(INP + 13); const int* idx_j2 = IP(INP + 14);
    const int* idx_jj  = IP(INP + 15); const int* idx_ji2 = IP(INP + 16);
    const int N  = in_sizes[INP + 0] / 3;
    const int El = in_sizes[INP + 3], Eg = in_sizes[INP + 5];
    const int T1 = in_sizes[INP + 10], T2 = in_sizes[INP + 15];
    int maxE = N;
    if (Eg > maxE) maxE = Eg; if (El > maxE) maxE = El;
    if (T1 > maxE) maxE = T1; if (T2 > maxE) maxE = T2;

    char* base = (char*)d_ws; size_t off = 0;
    auto alloc = [&](size_t b) -> char* { size_t a = (off + 255) & ~255ull; off = a + b; return base + a; };
    float* h      = (float*)alloc((size_t)N * D128 * 4);
    u16*   hb     = (u16*)  alloc((size_t)N * D128 * 2);
    u16*   rbfgb  = (u16*)  alloc((size_t)Eg * D128 * 2);
    u16*   rbflb  = (u16*)  alloc((size_t)El * D128 * 2);
    u16*   sbf1b  = (u16*)  alloc((size_t)T1 * D128 * 2);
    u16*   sbf2b  = (u16*)  alloc((size_t)T2 * D128 * 2);
    u16*   besg   = (u16*)  alloc((size_t)Eg * 64 * 2);
    u16*   besl   = (u16*)  alloc((size_t)El * 64 * 2);
    u16*   raw1   = (u16*)  alloc((size_t)T1 * 64 * 2);
    u16*   raw2   = (u16*)  alloc((size_t)T2 * 64 * 2);
    float* dl     = (float*)alloc((size_t)El * 4);
    float* sbfe   = (float*)alloc((size_t)El * 42 * 4);
    float* t1 = (float*)alloc((size_t)maxE * D128 * 4);
    float* t2 = (float*)alloc((size_t)maxE * D128 * 4);
    float* t3 = (float*)alloc((size_t)maxE * D128 * 4);
    float* t4 = (float*)alloc((size_t)maxE * D128 * 4);
    float* t5 = (float*)alloc((size_t)maxE * D128 * 4);
    float* tR = (float*)alloc((size_t)maxE * D128 * 4);
    u16* t1b = (u16*)alloc((size_t)maxE * D128 * 2);
    u16* t2b = (u16*)alloc((size_t)maxE * D128 * 2);
    u16* t3b = (u16*)alloc((size_t)maxE * D128 * 2);
    u16* tRb = (u16*)alloc((size_t)maxE * D128 * 2);
    u16* X3  = (u16*)alloc((size_t)maxE * 384 * 2);
    float* nodeSum = (float*)alloc((size_t)N * 4);
    char* wreg = alloc(32u << 20); size_t woff = 0;
    auto walloc = [&](size_t b) -> u16* { size_t a = (woff + 15) & ~15ull; woff = a + b; return (u16*)(wreg + a); };

    auto cdiv = [](long a, long b) { return (int)((a + b - 1) / b); };
    auto packW = [&](int leaf, int K, int Nn) -> u16* {
        int Kp = (K + 63) & ~63;
        u16* dst = walloc((size_t)Kp * Nn * 2);
        k_pack_w<<<cdiv((long)Kp * Nn, 256), 256, 0, stream>>>(FP(leaf), dst, K, Kp, Nn);
        return dst;
    };
    auto gemm = [&](const u16* Xp, const u16* Wp, const float* bias, int M, int K, int act,
                    const float* res, float* outF, u16* outB) {
        dim3 g(1, cdiv(M, 128));
        k_gemm<<<g, 256, 0, stream>>>(Xp, Wp, bias, res, outF, outB, M, D128, K, act);
    };
    auto resblk = [&](u16* W1, const float* b1, u16* W2, const float* b2,
                      float* x, u16* xb, int M) {
        gemm(xb, W1, b1, M, 128, 1, nullptr, nullptr, tRb);
        gemm(tRb, W2, b2, M, 128, 1, x, x, xb);
    };

    // ---------- precompute ----------
    (void)hipMemsetAsync(nodeSum, 0, (size_t)N * 4, stream);
    (void)hipMemsetAsync(d_out, 0, (size_t)out_size * 4, stream);
    k_embed<<<cdiv((long)N * D128, 256), 256, 0, stream>>>(FP(0), atom_z, h, hb, N);
    k_bessel<<<cdiv(Eg, 256), 256, 0, stream>>>(pos, src_g, dst_g, FP(1), nullptr, besg, Eg);
    k_bessel<<<cdiv(El, 256), 256, 0, stream>>>(pos, src_l, dst_l, FP(2), dl, besl, El);
    k_sbf_edge<<<cdiv(El, 256), 256, 0, stream>>>(dl, sbfe, El);
    k_tri<<<cdiv(T1, 256), 256, 0, stream>>>(pos, idx_i1, idx_j, idx_k, idx_kj, sbfe, raw1, T1);
    k_tri<<<cdiv(T2, 256), 256, 0, stream>>>(pos, idx_i2, idx_j1, idx_j2, idx_jj, sbfe, raw2, T2);
    { u16* W = packW(183, 16, 128); gemm(besg, W, FP(184), Eg, 64, 1, nullptr, nullptr, rbfgb); }
    { u16* W = packW(185, 16, 128); gemm(besl, W, FP(186), El, 64, 1, nullptr, nullptr, rbflb); }
    { u16* W = packW(187, 42, 128); gemm(raw1, W, FP(188), T1, 64, 1, nullptr, nullptr, sbf1b); }
    { u16* W = packW(189, 42, 128); gemm(raw2, W, FP(190), T2, 64, 1, nullptr, nullptr, sbf2b); }

    // ---------- layers ----------
    for (int L = 0; L < 3; ++L) {
        // ===== global message passing =====
        u16 *Wh  = packW(GLOB(L, 0), 128, 128), *Wlin = packW(GLOB(L, 2), 128, 128);
        u16 *Wm  = packW(GLOB(L, 3), 128, 128);
        u16 *W1a = packW(GLOB(L, 5), 128, 128), *W1b = packW(GLOB(L, 7), 128, 128);
        u16 *W2a = packW(GLOB(L, 9), 128, 128), *W2b = packW(GLOB(L, 11), 128, 128);
        u16 *W3a = packW(GLOB(L, 13), 128, 128), *W3b = packW(GLOB(L, 15), 128, 128);
        u16 *Wxe = packW(GLOB(L, 17), 384, 128);

        gemm(hb, Wh, FP(GLOB(L, 1)), N, 128, 1, nullptr, t1, t1b);                 // h' = mlp(h)
        k_concat3<<<cdiv((long)Eg * 48, 256), 256, 0, stream>>>(t1b, dst_g, src_g, rbfgb, X3, Eg);
        gemm(X3, Wxe, FP(GLOB(L, 18)), Eg, 384, 1, nullptr, t2, nullptr);          // xe
        gemm(rbfgb, Wlin, nullptr, Eg, 128, 0, nullptr, t3, nullptr);              // rbf@lin
        k_mul<<<cdiv((long)Eg * D128, 256), 256, 0, stream>>>(t2, t3, (long)Eg * D128);
        k_scatter_add<<<cdiv((long)Eg * D128, 256), 256, 0, stream>>>(t1, t2, dst_g, Eg);
        k_f2b<<<cdiv((long)N * D128, 256), 256, 0, stream>>>(t1b, t1, (long)N * D128);
        resblk(W1a, FP(GLOB(L, 6)), W1b, FP(GLOB(L, 8)), t1, t1b, N);              // res1
        gemm(t1b, Wm, FP(GLOB(L, 4)), N, 128, 1, h, h, hb);                        // + res_h
        resblk(W2a, FP(GLOB(L, 10)), W2b, FP(GLOB(L, 12)), h, hb, N);              // res2
        resblk(W3a, FP(GLOB(L, 14)), W3b, FP(GLOB(L, 16)), h, hb, N);              // res3

        // ===== local message passing =====
        u16 *LWh   = packW(LOC(L, 0), 128, 128);
        u16 *Wlr1  = packW(LOC(L, 2), 128, 128), *Wlr2 = packW(LOC(L, 3), 128, 128);
        u16 *Wlro  = packW(LOC(L, 4), 128, 128);
        u16 *Wji1  = packW(LOC(L, 5), 384, 128), *Wji2 = packW(LOC(L, 7), 128, 128);
        u16 *Wjj   = packW(LOC(L, 9), 128, 128), *Wkj  = packW(LOC(L, 11), 384, 128);
        u16 *Ws1a  = packW(LOC(L, 13), 128, 128), *Ws1b = packW(LOC(L, 15), 128, 128);
        u16 *Ws2a  = packW(LOC(L, 17), 128, 128), *Ws2b = packW(LOC(L, 19), 128, 128);
        u16 *Lr1a  = packW(LOC(L, 21), 128, 128), *Lr1b = packW(LOC(L, 23), 128, 128);
        u16 *Lr2a  = packW(LOC(L, 25), 128, 128), *Lr2b = packW(LOC(L, 27), 128, 128);
        u16 *Lr3a  = packW(LOC(L, 29), 128, 128), *Lr3b = packW(LOC(L, 31), 128, 128);
        u16 *Wy1   = packW(LOC(L, 35), 128, 128), *Wy2 = packW(LOC(L, 37), 128, 128);
        u16 *Wy3   = packW(LOC(L, 39), 128, 128);

        gemm(hb, LWh, FP(LOC(L, 1)), N, 128, 1, nullptr, t1, t1b);                 // h' = mlp(h)
        k_concat3<<<cdiv((long)El * 48, 256), 256, 0, stream>>>(t1b, dst_l, src_l, rbflb, X3, El);
        // s1 = mlp_sbf1(sbf1)
        gemm(sbf1b, Ws1a, FP(LOC(L, 14)), T1, 128, 1, nullptr, nullptr, t2b);
        gemm(t2b, Ws1b, FP(LOC(L, 16)), T1, 128, 1, nullptr, t4, nullptr);
        // m_kj = mlp_kj(m) * rbf@lin_rbf1
        gemm(X3, Wkj, FP(LOC(L, 12)), El, 384, 1, nullptr, t2, nullptr);
        gemm(rbflb, Wlr1, nullptr, El, 128, 0, nullptr, t3, nullptr);
        k_mul<<<cdiv((long)El * D128, 256), 256, 0, stream>>>(t2, t3, (long)El * D128);
        (void)hipMemsetAsync(t3, 0, (size_t)El * D128 * 4, stream);
        k_tri_gms<<<cdiv((long)T1 * D128, 256), 256, 0, stream>>>(t3, t2, t4, idx_kj, idx_ji1, T1);
        gemm(X3, Wji1, FP(LOC(L, 6)), El, 384, 1, t3, t2, t2b);                    // m
        // s2 = mlp_sbf2(sbf2)
        gemm(sbf2b, Ws2a, FP(LOC(L, 18)), T2, 128, 1, nullptr, nullptr, t1b);
        gemm(t1b, Ws2b, FP(LOC(L, 20)), T2, 128, 1, nullptr, t4, nullptr);
        // m_jj = mlp_jj(m) * rbf@lin_rbf2
        gemm(t2b, Wjj, FP(LOC(L, 10)), El, 128, 1, nullptr, t3, nullptr);
        gemm(rbflb, Wlr2, nullptr, El, 128, 0, nullptr, t5, nullptr);
        k_mul<<<cdiv((long)El * D128, 256), 256, 0, stream>>>(t3, t5, (long)El * D128);
        (void)hipMemsetAsync(t5, 0, (size_t)El * D128 * 4, stream);
        k_tri_gms<<<cdiv((long)T2 * D128, 256), 256, 0, stream>>>(t5, t3, t4, idx_jj, idx_ji2, T2);
        gemm(t2b, Wji2, FP(LOC(L, 8)), El, 128, 1, t5, t3, nullptr);               // m
        gemm(rbflb, Wlro, nullptr, El, 128, 0, nullptr, t5, nullptr);
        k_mul<<<cdiv((long)El * D128, 256), 256, 0, stream>>>(t3, t5, (long)El * D128);
        // h = segment_sum(m, dst)
        (void)hipMemsetAsync(t1, 0, (size_t)N * D128 * 4, stream);
        k_scatter_add<<<cdiv((long)El * D128, 256), 256, 0, stream>>>(t1, t3, dst_l, El);
        k_f2b<<<cdiv((long)N * D128, 256), 256, 0, stream>>>(t1b, t1, (long)N * D128);
        resblk(Lr1a, FP(LOC(L, 22)), Lr1b, FP(LOC(L, 24)), t1, t1b, N);            // res1
        gemm(t1b, LWh, FP(LOC(L, 1)), N, 128, 1, h, h, hb);                        // + res_h
        resblk(Lr2a, FP(LOC(L, 26)), Lr2b, FP(LOC(L, 28)), h, hb, N);              // res2
        resblk(Lr3a, FP(LOC(L, 30)), Lr3b, FP(LOC(L, 32)), h, hb, N);              // res3
        // y = y_mlp(h); node_sum += y@Wy + by
        gemm(hb, Wy1, FP(LOC(L, 36)), N, 128, 1, nullptr, nullptr, t1b);
        gemm(t1b, Wy2, FP(LOC(L, 38)), N, 128, 1, nullptr, nullptr, t2b);
        gemm(t2b, Wy3, FP(LOC(L, 40)), N, 128, 1, nullptr, t1, nullptr);
        k_dot<<<cdiv((long)N * 32, 256), 256, 0, stream>>>(t1, FP(LOC(L, 33)), FP(LOC(L, 34)), nodeSum, N);
    }

    k_mol_sum<<<cdiv(N, 256), 256, 0, stream>>>(nodeSum, batch, (float*)d_out, N);
    (void)ws_size;
#undef FP
#undef IP
}
